// OriLinearGNN_6846177869857
// MI455X (gfx1250) — compile-verified, matrix operand order checked by reading, not run on hardware
//
#include <hip/hip_runtime.h>
#include <hip/hip_bf16.h>

#define V_N   50000
#define E_N   200000
#define LN    128
#define SDIM  32
#define CN    10
#define XDIM  256      // 2*LN
#define NCOL  1024     // S*S
#define MU    0.9f

typedef __attribute__((ext_vector_type(16))) __bf16 v16bf;
typedef __attribute__((ext_vector_type(8)))  float  v8f;

// Branch-free tanh: prefer gfx1250 hardware v_tanh_f32.
__device__ __forceinline__ float fast_tanh(float x) {
#if __has_builtin(__builtin_amdgcn_tanhf)
    return __builtin_amdgcn_tanhf(x);
#else
    // 1 - 2/(e^{2x}+1); v_exp + v_rcp, no branches, saturates correctly.
    float e = __expf(2.0f * x);
    return 1.0f - 2.0f * __builtin_amdgcn_rcpf(e + 1.0f);
#endif
}

// ---------------------------------------------------------------- zero scratch
__global__ void zero_kernel(float* __restrict__ p, int n) {
    int i = blockIdx.x * blockDim.x + threadIdx.x;
    if (i < n) p[i] = 0.0f;
}

// ------------------------------------------------- repack Wxi -> bf16 B-fragments
// Fragment order: [tile t (64)][kstep kk (8)][lane (32)][slot (16)]
// 16-bit B 32x16 layout mirrors A: lane = h*16 + n (n = column),
// K(h,r,p) = (r>=4)*16 + h*8 + (r&3)*2 + p,  slot = r*2+p.
__global__ void repack_wxi_kernel(const float* __restrict__ Wxi,
                                  __bf16* __restrict__ out) {
    int idx  = blockIdx.x * 256 + threadIdx.x;      // 0 .. 262143
    int t    = idx >> 12;
    int kk   = (idx >> 9) & 7;
    int lane = (idx >> 4) & 31;
    int slot = idx & 15;
    int r = slot >> 1, p = slot & 1;
    int h = lane >> 4, n = lane & 15;
    int K   = kk * 32 + ((r >= 4) ? 16 : 0) + h * 8 + (r & 3) * 2 + p;
    int col = t * 16 + n;
    out[idx] = (__bf16)Wxi[K * NCOL + col];
}

// ---------------------------------------------- pass 1: H1 = segment_sum(b)
// thread layout: edge = tid>>5 (wave-uniform feat addr -> scalar loads), s = lane
__global__ void edge_b_kernel(const float* __restrict__ feat,
                              const int*   __restrict__ xnode,
                              const float* __restrict__ Wrou,
                              const float* __restrict__ brou,
                              float* __restrict__ H1) {
    int el = threadIdx.x >> 5;
    int s  = threadIdx.x & 31;
    int e  = blockIdx.x * 8 + el;
    int u  = xnode[e];
    const float* fr = feat + (long)u * LN;
    float acc = brou[s];
#pragma unroll 8
    for (int k = 0; k < LN; ++k) acc += fr[k] * Wrou[k * SDIM + s];
    atomicAdd(&H1[(long)u * SDIM + s], fast_tanh(acc));
}

// -------------------------------- pass 2 (hot): He = A@H1[u] + b ; H2 += scatter
__global__ __launch_bounds__(256, 2)
void edge_heavy_kernel(const float*  __restrict__ feat,
                       const int*    __restrict__ xnode,
                       const int*    __restrict__ xneis,
                       const float*  __restrict__ dg,
                       const __bf16* __restrict__ WxiB,
                       const float*  __restrict__ bxi,
                       const float*  __restrict__ Wrou,
                       const float*  __restrict__ brou,
                       const float*  __restrict__ H1,
                       float*        __restrict__ H2) {
    __shared__ __bf16 ldsA[8][32][16];   // X tile pre-swizzled into A-fragments (8 KB)
    __shared__ float  Xf[16][LN];        // node-feature half, f32, for b recompute (8 KB)
    __shared__ float  H1g[16][SDIM];     // gathered H1 rows, PRE-SCALED by sc_e (2 KB)
    __shared__ float  He[16][SDIM];      // per-block accumulators (2 KB)
    __shared__ float  bxis[NCOL];        // bias (4 KB)
    __shared__ int    us[16];

    const int tid = threadIdx.x;
    const int eb  = blockIdx.x * 16;

    if (tid < 16) us[tid] = xnode[eb + tid];
    for (int i = tid; i < NCOL; i += 256) bxis[i] = bxi[i];

    // ---- stage X tile: e = tid>>4, 16 consecutive k per thread
    {
        int e  = tid >> 4;
        int k0 = (tid & 15) * 16;
        int un = xnode[eb + e], vn = xneis[eb + e];
        const float* fu = feat + (long)un * LN;
        const float* fv = feat + (long)vn * LN;
#pragma unroll
        for (int q = 0; q < 16; ++q) {
            int k = k0 + q;
            float val = (k < LN) ? fu[k] : fv[k - LN];
            if (k < LN) Xf[e][k] = val;
            int kt = k >> 5, kl = k & 31;
            int h  = ((kl & 15) >= 8) ? 1 : 0;
            int r  = ((kl >= 16) ? 4 : 0) + ((kl & 7) >> 1);
            int p  = kl & 1;
            ldsA[kt][h * 16 + e][r * 2 + p] = (__bf16)val;
        }
    }
    // ---- gather H1 rows pre-scaled by (MU/S)/deg  (sc_e factors out of A@H1),
    //      and zero He
    const float scale_base = MU / (float)SDIM;
    for (int i = tid; i < 16 * SDIM; i += 256) {
        int ee = i >> 5, j = i & 31;
        float sc = scale_base * __builtin_amdgcn_rcpf(dg[eb + ee]);
        H1g[ee][j] = H1[(long)xnode[eb + ee] * SDIM + j] * sc;
        He[ee][j]  = 0.0f;
    }
    __syncthreads();

    // ---- WMMA main loop: each wave owns 8 column tiles (4 pairs sharing i_row)
    const int lane = tid & 31;
    const int wave = tid >> 5;
    const int h    = lane >> 4;
    const int n    = lane & 15;

    // hoist the 8 A-fragments (shared by all tiles of this wave)
    v16bf afrag[8];
#pragma unroll
    for (int kk = 0; kk < 8; ++kk)
        afrag[kk] = *(const v16bf*)(&ldsA[kk][lane][0]);

#pragma unroll
    for (int tp = 0; tp < 4; ++tp) {          // tile pair; i_row fixed per pair
        const int i_row = wave * 4 + tp;
        float red[8];
#pragma unroll
        for (int r = 0; r < 8; ++r) red[r] = 0.0f;

#pragma unroll
        for (int half = 0; half < 2; ++half) {
            const int t = wave * 8 + tp * 2 + half;
            // two independent accumulator chains for matrix-pipe ILP
            v8f acc0 = {0.f, 0.f, 0.f, 0.f, 0.f, 0.f, 0.f, 0.f};
            v8f acc1 = {0.f, 0.f, 0.f, 0.f, 0.f, 0.f, 0.f, 0.f};
#pragma unroll
            for (int kk = 0; kk < 8; kk += 2) {
                v16bf b0 = *(const v16bf*)(WxiB + (((t * 8 + kk)     * 32 + lane) << 4));
                v16bf b1 = *(const v16bf*)(WxiB + (((t * 8 + kk + 1) * 32 + lane) << 4));
                acc0 = __builtin_amdgcn_wmma_f32_16x16x32_bf16(
                           false, afrag[kk],     false, b0, (short)0, acc0, false, false);
                acc1 = __builtin_amdgcn_wmma_f32_16x16x32_bf16(
                           false, afrag[kk + 1], false, b1, (short)0, acc1, false, false);
            }
            const int   jj = (half << 4) + n;       // A-matrix col index (S dim)
            const float bx = bxis[t * 16 + n];
            // tanh + fma only: per-edge scale is pre-folded into H1g
#pragma unroll
            for (int r = 0; r < 8; ++r) {
                int ee = r + h * 8;                 // local edge
                red[r] += fast_tanh(acc0[r] + acc1[r] + bx) * H1g[ee][jj];
            }
        }
        // reduce across the 16 lanes of each half-wave, then one LDS atomic per row
#pragma unroll
        for (int r = 0; r < 8; ++r) {
            float s = red[r];
            s += __shfl_xor(s, 1, 16);
            s += __shfl_xor(s, 2, 16);
            s += __shfl_xor(s, 4, 16);
            s += __shfl_xor(s, 8, 16);
            if (n == 0) atomicAdd(&He[r + h * 8][i_row], s);
        }
    }
    __syncthreads();

    // ---- epilogue: add recomputed b, scatter to H2
#pragma unroll
    for (int pass = 0; pass < 2; ++pass) {
        int ee = (tid >> 5) + pass * 8;
        int s  = tid & 31;
        float acc2 = brou[s];
#pragma unroll 8
        for (int k = 0; k < LN; ++k) acc2 += Xf[ee][k] * Wrou[k * SDIM + s];
        float heval = He[ee][s] + fast_tanh(acc2);
        atomicAdd(&H2[(long)us[ee] * SDIM + s], heval);
    }
}

// ------------------------------------------- output head + log_softmax
__global__ void out_kernel(const float* __restrict__ H2,
                           const float* __restrict__ Wout,
                           const float* __restrict__ bout,
                           float* __restrict__ out) {
    int v = blockIdx.x * blockDim.x + threadIdx.x;
    if (v >= V_N) return;
    float hrow[SDIM];
#pragma unroll
    for (int s = 0; s < SDIM; ++s) hrow[s] = H2[(long)v * SDIM + s];
    float lg[CN];
    float m = -1e30f;
#pragma unroll
    for (int c = 0; c < CN; ++c) {
        float a = bout[c];
#pragma unroll
        for (int s = 0; s < SDIM; ++s) a += hrow[s] * Wout[s * CN + c];
        lg[c] = a;
        m = fmaxf(m, a);
    }
    float se = 0.0f;
#pragma unroll
    for (int c = 0; c < CN; ++c) se += __expf(lg[c] - m);
    float lse = m + __logf(se);
#pragma unroll
    for (int c = 0; c < CN; ++c) out[(long)v * CN + c] = lg[c] - lse;
}

extern "C" void kernel_launch(void* const* d_in, const int* in_sizes, int n_in,
                              void* d_out, int out_size, void* d_ws, size_t ws_size,
                              hipStream_t stream) {
    const float* feat  = (const float*)d_in[0];
    const int*   xnode = (const int*)  d_in[1];
    const int*   xneis = (const int*)  d_in[2];
    const float* dg    = (const float*)d_in[3];
    const float* Wxi   = (const float*)d_in[4];
    const float* bxi   = (const float*)d_in[5];
    const float* Wrou  = (const float*)d_in[6];
    const float* brou  = (const float*)d_in[7];
    const float* Wout  = (const float*)d_in[8];
    const float* bout  = (const float*)d_in[9];

    float*  H1   = (float*)d_ws;                       // V*S floats
    float*  H2   = H1 + (size_t)V_N * SDIM;            // V*S floats
    __bf16* WxiB = (__bf16*)(H2 + (size_t)V_N * SDIM); // 256*1024 bf16 (32B aligned)
    float*  out  = (float*)d_out;

    const int hTot = 2 * V_N * SDIM;
    zero_kernel<<<(hTot + 255) / 256, 256, 0, stream>>>(H1, hTot);
    repack_wxi_kernel<<<(64 * 8 * 32 * 16) / 256, 256, 0, stream>>>(Wxi, WxiB);
    edge_b_kernel<<<E_N / 8, 256, 0, stream>>>(feat, xnode, Wrou, brou, H1);
    edge_heavy_kernel<<<E_N / 16, 256, 0, stream>>>(feat, xnode, xneis, dg, WxiB,
                                                    bxi, Wrou, brou, H1, H2);
    out_kernel<<<(V_N + 255) / 256, 256, 0, stream>>>(H2, Wout, bout, out);
}
